// RNN_model_50946902065494
// MI455X (gfx1250) — matrix operand order repaired and checked by reading, not compiled, not run on hardware
//
#include <hip/hip_runtime.h>
#include <hip/hip_bf16.h>

// ---------------------------------------------------------------------------
// Problem constants (match reference)
// ---------------------------------------------------------------------------
static constexpr int T_   = 2048;
static constexpr int B_   = 512;
static constexpr int IN_  = 27;
static constexpr int H_   = 64;
static constexpr int G4_  = 4 * H_;       // 256 gate columns
static constexpr int OUT_ = 26;
static constexpr int FCK_ = H_ + OUT_;    // 90
static constexpr float EPS_ = 1e-5f;

// WMMA tiling
static constexpr int KX_   = 32;          // x K padded 27 -> 32
static constexpr int KTOT_ = KX_ + H_;    // 96 combined K
static constexpr int KP_   = 120;         // padded LDS row stride (halves): 240B,
                                          // 60 dwords -> conflict-free & 16B aligned
static constexpr int BPW_  = 16;          // batch rows per workgroup (M tile)
static constexpr int GP_   = G4_ + 4;     // gate LDS row stride (floats)

typedef __attribute__((ext_vector_type(16))) _Float16 v16h;
typedef __attribute__((ext_vector_type(8)))  _Float16 v8h;
typedef __attribute__((ext_vector_type(8)))  float    v8f;

__device__ __forceinline__ v16h combine8(v8h lo, v8h hi) {
    v16h r;
#pragma unroll
    for (int i = 0; i < 8; ++i) { r[i] = lo[i]; r[i + 8] = hi[i]; }
    return r;
}

// ---- fast transcendentals: prefer gfx1250 v_tanh_f32, else rcp+exp ---------
__device__ __forceinline__ float tanh_f(float x) {
#if __has_builtin(__builtin_amdgcn_tanhf)
    return __builtin_amdgcn_tanhf(x);
#elif __has_builtin(__builtin_amdgcn_tanh_f32)
    return __builtin_amdgcn_tanh_f32(x);
#else
    // tanh(x) = 1 - 2/(exp(2x)+1), single v_exp + v_rcp
    return 1.0f - 2.0f * __builtin_amdgcn_rcpf(__expf(2.0f * x) + 1.0f);
#endif
}
__device__ __forceinline__ float sigmoid_f(float x) {
#if __has_builtin(__builtin_amdgcn_tanhf) || __has_builtin(__builtin_amdgcn_tanh_f32)
    // sigmoid(x) = 0.5*tanh(x/2) + 0.5  (one TRANS op + FMA)
    return fmaf(0.5f, tanh_f(0.5f * x), 0.5f);
#else
    return __builtin_amdgcn_rcpf(1.0f + __expf(-x));
#endif
}

__device__ __forceinline__ void wait_asynccnt0() {
#if __has_builtin(__builtin_amdgcn_s_wait_asynccnt)
    __builtin_amdgcn_s_wait_asynccnt(0);
#else
    asm volatile("s_wait_asynccnt 0x0" ::: "memory");
#endif
}

// ---------------------------------------------------------------------------
// Kernel A: batched LSTM recurrence.
// grid.x = B/16 = 32 workgroups; block = 512 threads (16 waves of 32).
// Wave w computes gate columns [16w, 16w+16) for the workgroup's 16 batch rows
// via 3x v_wmma_f32_16x16x32_f16 (K = 96 fused x|h), bias pre-seeded in C.
// ---------------------------------------------------------------------------
__global__ __launch_bounds__(512, 1)
void lstm_recurrent_kernel(const float* __restrict__ x,      // [T,B,IN]
                           const float* __restrict__ W_ih,   // [4H,IN]
                           const float* __restrict__ W_hh,   // [4H,H]
                           const float* __restrict__ b_ih,   // [4H]
                           const float* __restrict__ b_hh,   // [4H]
                           float* __restrict__ h_seq)        // [T,B,H] workspace
{
    __shared__ _Float16 sW[G4_][KP_];    // fused weights, column-major: sW[n][k]
    __shared__ _Float16 sXH[BPW_][KP_];  // A staging: [row][k], k<32 = x, 32..95 = h
    __shared__ float    sGate[BPW_][GP_];
    __shared__ float    sBias[G4_];

    const int tid  = threadIdx.x;
    const int wave = tid >> 5;
    const int lane = tid & 31;
    const int hi16 = (lane >> 4) & 1;    // 0 for lanes 0-15, 1 for 16-31
    const int b0   = blockIdx.x * BPW_;

    // ---- one-time: fused f16 weights (column-major) + bias into LDS --------
    {
        const int n    = tid & 255;
        const int half = tid >> 8;       // 0 or 1
        const int k0   = half * (KTOT_ / 2);
#pragma unroll 4
        for (int k = k0; k < k0 + KTOT_ / 2; ++k) {
            float v;
            if (k < IN_)        v = W_ih[n * IN_ + k];
            else if (k < KX_)   v = 0.0f;
            else                v = W_hh[n * H_ + (k - KX_)];
            sW[n][k] = (_Float16)v;
        }
        if (tid < G4_) sBias[tid] = b_ih[tid] + b_hh[tid];
        // zero the h part of the A staging (h0 = 0)
        for (int idx = tid; idx < BPW_ * H_; idx += 512) {
            sXH[idx >> 6][KX_ + (idx & 63)] = (_Float16)0.0f;
        }
    }
    __syncthreads();

    // ---- hoist B operand (constant across t) + bias into registers ---------
    // B k order used here (contiguous, same as A): lane n<16 holds K=0..15 of column n; lanes 16-31 hold
    // K=16..31 of column n-16.  16 contiguous halves per lane.
    const int ncol = wave * 16 + (lane & 15);
    v16h Bop[3];
#pragma unroll
    for (int kc = 0; kc < 3; ++kc) {
        const int k0 = kc * 32 + hi16 * 16;
        v8h lo = *(const v8h*)&sW[ncol][k0];
        v8h hi = *(const v8h*)&sW[ncol][k0 + 8];
        Bop[kc] = combine8(lo, hi);
    }
    const float biasc = sBias[ncol];     // bias is uniform across accumulator rows

    // ---- persistent cell state: 2 elements per thread ----------------------
    const int eb = tid >> 5;             // batch row 0..15 (== wave)
    const int ej = lane * 2;             // h column base 0..62
    float cst[2] = {0.0f, 0.0f};

    for (int t = 0; t < T_; ++t) {
        // stage x_t tile (16 rows x 32 padded cols), one element per thread
        {
            const int r = tid >> 5;
            const int k = tid & 31;
            float v = 0.0f;
            if (k < IN_)
                v = x[((size_t)t * B_ + (b0 + r)) * IN_ + k];
            sXH[r][k] = (_Float16)v;
        }
        if (t + 1 < T_) {
            __builtin_prefetch(&x[((size_t)(t + 1) * B_ + b0) * IN_], 0, 0);
        }
        __syncthreads();

        // ---- 16x16 gate tile: D = A(16x96) x B(96x16) + bias ---------------
        v8f acc;
#pragma unroll
        for (int r = 0; r < 8; ++r) acc[r] = biasc;   // seed C with bias splat
        const int arow = lane & 15;
#pragma unroll
        for (int kc = 0; kc < 3; ++kc) {
            // A k order used here (contiguous, same as B): halves 0..7  = K 16*hi16 + 0..7
            //                              halves 8..15 = K 16*hi16 + 8..15
            const int kb = kc * 32 + hi16 * 16;
            v8h lo = *(const v8h*)&sXH[arow][kb];
            v8h hi = *(const v8h*)&sXH[arow][kb + 8];
            v16h A = combine8(lo, hi);
            acc = __builtin_amdgcn_wmma_f32_16x16x32_f16(
                false, A, false, Bop[kc], (short)0, acc, false, false);
        }

        // scatter accumulator to LDS: (M = r + 8*hi16, N = ncol)
#pragma unroll
        for (int r = 0; r < 8; ++r) {
            sGate[r + hi16 * 8][ncol] = acc[r];
        }
        __syncthreads();

        // ---- elementwise LSTM cell update (2 h-values per thread) ----------
        float hnew[2];
#pragma unroll
        for (int i = 0; i < 2; ++i) {
            const int j = ej + i;
            const float ig = sigmoid_f(sGate[eb][j]);
            const float fg = sigmoid_f(sGate[eb][H_ + j]);
            const float gg = tanh_f   (sGate[eb][2 * H_ + j]);
            const float og = sigmoid_f(sGate[eb][3 * H_ + j]);
            cst[i] = fg * cst[i] + ig * gg;
            hnew[i] = og * tanh_f(cst[i]);
            sXH[eb][KX_ + j] = (_Float16)hnew[i];   // feed next step's A
        }
        // coalesced f32 store of h_new for the T-parallel post pass
        {
            float2 h2 = make_float2(hnew[0], hnew[1]);
            *(float2*)&h_seq[((size_t)t * B_ + (b0 + eb)) * H_ + ej] = h2;
        }
        __syncthreads();
    }
}

// ---------------------------------------------------------------------------
// Kernel B: per-timestep BatchNorm + dropout + batch max-pool + FC.
// grid.x = T blocks; block = 256 threads = 64 cols x 4 batch segments.
// The full 128 KB h[t] tile is brought to LDS once via async-to-LDS loads
// (ASYNCcnt path), then both passes (stats, max) read LDS only.
// ---------------------------------------------------------------------------
__global__ __launch_bounds__(256, 2)
void postproc_kernel(const float* __restrict__ h_seq,   // [T,B,H]
                     const float* __restrict__ pg,      // [T,OUT]
                     const float* __restrict__ gamma,
                     const float* __restrict__ beta,
                     const float* __restrict__ W_fc,    // [OUT, H+OUT]
                     const float* __restrict__ b_fc,
                     const float* __restrict__ mask,    // [B,H]
                     float* __restrict__ out)           // [T,1,OUT]
{
    __shared__ float sH[B_ * H_];        // 128 KB h[t] tile
    __shared__ float sSum[4][H_];
    __shared__ float sSq[4][H_];
    __shared__ float sA[H_], sB[H_];
    __shared__ float sPool[H_];

    const int t   = blockIdx.x;
    const int tid = threadIdx.x;
    const int col = tid & 63;
    const int seg = tid >> 6;
    const float* gsrc = h_seq + (size_t)t * B_ * H_;

    // ---- async copy of the whole tile into LDS (b128 chunks) ---------------
    {
        const unsigned lbase = (unsigned)(unsigned long long)(uintptr_t)&sH[0];
        constexpr int NCHUNK = (B_ * H_ * 4) / 16;       // 8192 x 16B
#pragma unroll 4
        for (int k = 0; k < NCHUNK / 256; ++k) {         // 32 per thread
            const unsigned off = (unsigned)(tid + k * 256) * 16u;
            asm volatile("global_load_async_to_lds_b128 %0, %1, %2"
                         :: "v"(lbase + off), "v"(off), "s"(gsrc)
                         : "memory");
        }
        wait_asynccnt0();
    }
    __syncthreads();

    // pass 1: batch statistics (biased variance, matches jnp.mean((h-mu)^2))
    float sum = 0.0f, sq = 0.0f;
    for (int b = seg; b < B_; b += 4) {
        const float v = sH[(b << 6) + col];
        sum += v;
        sq  += v * v;
    }
    sSum[seg][col] = sum;
    sSq[seg][col]  = sq;
    __syncthreads();
    if (tid < H_) {
        float s = sSum[0][tid] + sSum[1][tid] + sSum[2][tid] + sSum[3][tid];
        float q = sSq[0][tid]  + sSq[1][tid]  + sSq[2][tid]  + sSq[3][tid];
        const float mean = s * (1.0f / B_);
        const float var  = q * (1.0f / B_) - mean * mean;
        const float a    = gamma[tid] * rsqrtf(var + EPS_);
        sA[tid] = a;
        sB[tid] = beta[tid] - mean * a;
    }
    __syncthreads();

    // pass 2: normalized + dropout-mask + max over batch
    float m = -3.402823466e+38f;
    {
        const float a = sA[col], bb = sB[col];
        for (int b = seg; b < B_; b += 4) {
            const float v  = sH[(b << 6) + col];
            const float hd = fmaf(v, a, bb) * mask[(size_t)b * H_ + col];
            m = fmaxf(m, hd);
        }
    }
    sSum[seg][col] = m;
    __syncthreads();
    if (tid < H_) {
        sPool[tid] = fmaxf(fmaxf(sSum[0][tid], sSum[1][tid]),
                           fmaxf(sSum[2][tid], sSum[3][tid]));
    }
    __syncthreads();

    // FC: y = [pooled | pg_t] @ W_fc^T + b_fc
    if (tid < OUT_) {
        float y = b_fc[tid];
        const float* wrow = W_fc + tid * FCK_;
#pragma unroll 8
        for (int k = 0; k < H_; ++k)   y = fmaf(wrow[k], sPool[k], y);
#pragma unroll
        for (int k = 0; k < OUT_; ++k) y = fmaf(wrow[H_ + k], pg[(size_t)t * OUT_ + k], y);
        out[(size_t)t * OUT_ + tid] = y;
    }
}

// ---------------------------------------------------------------------------
// Host-side launcher
// ---------------------------------------------------------------------------
extern "C" void kernel_launch(void* const* d_in, const int* in_sizes, int n_in,
                              void* d_out, int out_size, void* d_ws, size_t ws_size,
                              hipStream_t stream) {
    const float* x      = (const float*)d_in[0];   // [T,B,IN]
    const float* pg     = (const float*)d_in[1];   // [T,OUT]
    const float* W_ih   = (const float*)d_in[2];   // [4H,IN]
    const float* W_hh   = (const float*)d_in[3];   // [4H,H]
    const float* b_ih   = (const float*)d_in[4];   // [4H]
    const float* b_hh   = (const float*)d_in[5];   // [4H]
    const float* gamma  = (const float*)d_in[6];   // [H]
    const float* beta   = (const float*)d_in[7];   // [H]
    const float* W_fc   = (const float*)d_in[8];   // [OUT,H+OUT]
    const float* b_fc   = (const float*)d_in[9];   // [OUT]
    const float* mask   = (const float*)d_in[10];  // [B,H]
    float* out          = (float*)d_out;           // [T,1,OUT]

    float* h_seq = (float*)d_ws;                   // [T,B,H] f32 = 256 MB

    lstm_recurrent_kernel<<<B_ / BPW_, 512, 0, stream>>>(
        x, W_ih, W_hh, b_ih, b_hh, h_seq);

    postproc_kernel<<<T_, 256, 0, stream>>>(
        h_seq, pg, gamma, beta, W_fc, b_fc, mask, out);
}
